// ScaledSelfAttentionHead_7447473291399
// MI455X (gfx1250) — matrix operand order repaired and checked by reading, not compiled, not run on hardware
//
#include <hip/hip_runtime.h>
#include <hip/hip_bf16.h>

#define BATCH  8
#define SEQ    2048
#define EMB    1024
#define HEAD_D 128
#define SCALE  0.08838834764831845f   // 1/sqrt(128)

typedef __attribute__((ext_vector_type(16))) __bf16 v16bf;
typedef __attribute__((ext_vector_type(8)))  __bf16 v8bf;
typedef __attribute__((ext_vector_type(4)))  __bf16 v4bf;
typedef __attribute__((ext_vector_type(8)))  float  v8f;
typedef int v4i_g __attribute__((vector_size(16)));   // GCC-style, matches builtin sig

// ---- gfx1250 async global->LDS path (guarded; falls back to uint4 copies) --
#if defined(__AMDGCN__) && __has_builtin(__builtin_amdgcn_global_load_async_to_lds_b128) && __has_builtin(__builtin_amdgcn_s_wait_asynccnt)
#define HAVE_ASYNC_LDS 1
#else
#define HAVE_ASYNC_LDS 0
#endif

__device__ __forceinline__ void g2s_b128(__bf16* dst, const __bf16* src) {
#if HAVE_ASYNC_LDS
  __builtin_amdgcn_global_load_async_to_lds_b128(
      (__attribute__((address_space(1))) v4i_g*)(src),
      (__attribute__((address_space(3))) v4i_g*)(dst), 0, 0);
#else
  *(uint4*)dst = *(const uint4*)src;
#endif
}

__device__ __forceinline__ void async_wait0() {
#if HAVE_ASYNC_LDS
  __builtin_amdgcn_s_wait_asynccnt(0);
#endif
}

__device__ __forceinline__ v16bf make16(v8bf lo, v8bf hi) {
  v16bf r;
#pragma unroll
  for (int i = 0; i < 8; ++i) { r[i] = lo[i]; r[i + 8] = hi[i]; }
  return r;
}

// A operand: 16x32 bf16, rows rowBase..+15, storage row-major [row][k] ld elems.
// Lanes 0-15: K {0..7, 16..23}; lanes 16-31: K {8..15, 24..31}  (ISA 7.12.2).
__device__ __forceinline__ v16bf load_a_frag(const __bf16* base, int ld, int rowBase, int lane) {
  const int row = rowBase + (lane & 15);
  const int k0  = (lane >> 4) << 3;                 // 0 or 8
  const __bf16* p = base + row * ld + k0;
  return make16(*(const v8bf*)p, *(const v8bf*)(p + 16));
}

// B operand: 32x16 bf16 (KxN), storage N-major [n][k] ld elems.
__device__ __forceinline__ v16bf load_b_frag(const __bf16* bt, int ld, int colBase, int lane) {
  const int col = colBase + (lane & 15);
  const int k0  = (lane >> 4) << 3;                 // 0 or 8
  const __bf16* p = bt + col * ld + k0;
  return make16(*(const v8bf*)p, *(const v8bf*)(p + 16));
}

__device__ __forceinline__ v8f wmma_bf16(v16bf a, v16bf b, v8f c) {
  return __builtin_amdgcn_wmma_f32_16x16x32_bf16(false, a, false, b, (short)0, c, false, false);
}

// ---------------------------------------------------------------------------
// Kernel 1: q/k/v = X @ W + b  (blockIdx.y selects projection), software-
// pipelined: next K-slab loads to registers while WMMA consumes current LDS.
// q,k stored [b*T + t][h] bf16 ; v stored transposed vT[b][h][t] bf16.
// ---------------------------------------------------------------------------
__global__ void __launch_bounds__(256) qkv_proj_kernel(
    const float* __restrict__ X,
    const float* __restrict__ Wq, const float* __restrict__ bq,
    const float* __restrict__ Wk, const float* __restrict__ bk,
    const float* __restrict__ Wv, const float* __restrict__ bv,
    __bf16* __restrict__ qb, __bf16* __restrict__ kb, __bf16* __restrict__ vT)
{
  constexpr int LDT = 40, LDC = 136;
  __shared__ __align__(16) __bf16 As[128 * LDT];   // X tile [m][k]
  __shared__ __align__(16) __bf16 Bs[128 * LDT];   // W tile [n][k] (transposed)
  __shared__ __align__(16) __bf16 Cs[128 * LDC];   // output transpose staging
  const int tid  = threadIdx.x;
  const int lane = tid & 31;
  const int wid  = tid >> 5;
  const int p    = blockIdx.y;
  const int bt0  = blockIdx.x * 128;

  const float* W    = (p == 0) ? Wq : (p == 1) ? Wk : Wv;
  const float* bias = (p == 0) ? bq : (p == 1) ? bk : bv;

  const int xr = tid >> 3, xc = (tid & 7) << 2;     // X staging coords
  const int wk = tid >> 5, wn = (tid & 31) << 2;    // W staging coords

  float4 xv[4], wv4[4];
  auto loadSlab = [&](int k0) {
#pragma unroll
    for (int i = 0; i < 4; ++i)
      xv[i] = *(const float4*)&X[(size_t)(bt0 + xr + i * 32) * EMB + k0 + xc];
#pragma unroll
    for (int i = 0; i < 4; ++i)
      wv4[i] = *(const float4*)&W[(size_t)(k0 + wk + i * 8) * HEAD_D + wn];
  };
  auto storeSlab = [&]() {
#pragma unroll
    for (int i = 0; i < 4; ++i) {
      v4bf t;
      t[0] = (__bf16)xv[i].x; t[1] = (__bf16)xv[i].y;
      t[2] = (__bf16)xv[i].z; t[3] = (__bf16)xv[i].w;
      *(v4bf*)&As[(xr + i * 32) * LDT + xc] = t;
      const int k = wk + i * 8;
      Bs[(wn + 0) * LDT + k] = (__bf16)wv4[i].x;
      Bs[(wn + 1) * LDT + k] = (__bf16)wv4[i].y;
      Bs[(wn + 2) * LDT + k] = (__bf16)wv4[i].z;
      Bs[(wn + 3) * LDT + k] = (__bf16)wv4[i].w;
    }
  };

  v8f acc[8] = {};
  loadSlab(0);
  storeSlab();
  for (int k0 = 0; k0 < EMB; k0 += 32) {
    __syncthreads();                                 // current tile visible
    if (k0 + 32 < EMB) loadSlab(k0 + 32);            // overlap with WMMA below
    const v16bf a = load_a_frag(As, LDT, wid * 16, lane);
#pragma unroll
    for (int nt = 0; nt < 8; ++nt) {
      const v16bf b = load_b_frag(Bs, LDT, nt * 16, lane);
      acc[nt] = wmma_bf16(a, b, acc[nt]);
    }
    __syncthreads();                                 // all consumers done
    if (k0 + 32 < EMB) storeSlab();
  }

  const int b = bt0 / SEQ;
  if (p == 2) {
    // lane holds 8 consecutive t for one h -> single 16B store per tile
    const int tloc0 = (bt0 % SEQ) + wid * 16 + ((lane >> 4) << 3);
#pragma unroll
    for (int nt = 0; nt < 8; ++nt) {
      const int col = nt * 16 + (lane & 15);
      const float bb = bias[col];
      v8bf t;
#pragma unroll
      for (int v = 0; v < 8; ++v) t[v] = (__bf16)(acc[nt][v] + bb);
      *(v8bf*)&vT[(size_t)b * HEAD_D * SEQ + (size_t)col * SEQ + tloc0] = t;
    }
  } else {
    __bf16* o = (p == 0) ? qb : kb;
#pragma unroll
    for (int nt = 0; nt < 8; ++nt) {
      const int col = nt * 16 + (lane & 15);
      const float bb = bias[col];
#pragma unroll
      for (int v = 0; v < 8; ++v) {
        const int row = wid * 16 + v + ((lane >> 4) << 3);
        Cs[row * LDC + col] = (__bf16)(acc[nt][v] + bb);
      }
    }
    __syncthreads();
    for (int i = tid; i < 128 * 16; i += 256) {      // coalesced 16B stores
      const int r = i >> 4, c = (i & 15) << 3;
      *(uint4*)&o[(size_t)(bt0 + r) * HEAD_D + c] = *(const uint4*)&Cs[r * LDC + c];
    }
  }
}

// ---------------------------------------------------------------------------
// Kernel 2: den[b][s] = sum_{t>=s} exp(scale * q[t].k[s])
// One block per (batch, lower-triangular 128x128 tile pair).
// ---------------------------------------------------------------------------
__global__ void __launch_bounds__(256) colsum_kernel(
    const __bf16* __restrict__ qb, const __bf16* __restrict__ kb,
    float* __restrict__ den)
{
  constexpr int LDQ = 136;                   // 128 + 8 pad
  __shared__ __align__(16) __bf16 Qs[128 * LDQ];
  __shared__ __align__(16) __bf16 Ks[128 * LDQ];
  __shared__ float dsum[128];
  const int tid  = threadIdx.x;
  const int lane = tid & 31;
  const int wid  = tid >> 5;
  const int b    = blockIdx.y;

  int pair = blockIdx.x;                     // decode triangular (ti, sj)
  int ti = 0;
  while ((ti + 1) * (ti + 2) / 2 <= pair) ++ti;
  const int sj = pair - ti * (ti + 1) / 2;
  const int t0 = ti * 128, s0 = sj * 128;

  {
    const __bf16* qsrc = qb + (size_t)(b * SEQ + t0) * HEAD_D;
    const __bf16* ksrc = kb + (size_t)(b * SEQ + s0) * HEAD_D;
    for (int i = tid; i < 128 * 16; i += 256) {
      const int r = i >> 4, c = (i & 15) << 3;
      g2s_b128(&Qs[r * LDQ + c], &qsrc[r * HEAD_D + c]);
      g2s_b128(&Ks[r * LDQ + c], &ksrc[r * HEAD_D + c]);
    }
  }
  if (tid < 128) dsum[tid] = 0.f;
  async_wait0();
  __syncthreads();

  v16bf aq[4];                               // Q A-frags invariant across nt
#pragma unroll
  for (int h4 = 0; h4 < 4; ++h4)
    aq[h4] = load_a_frag(Qs + h4 * 32, LDQ, wid * 16, lane);

#pragma unroll
  for (int nt = 0; nt < 8; ++nt) {
    v8f d = {};
#pragma unroll
    for (int h4 = 0; h4 < 4; ++h4) {
      const v16bf bfr = load_b_frag(Ks + h4 * 32, LDQ, nt * 16, lane);
      d = wmma_bf16(aq[h4], bfr, d);
    }
    const int sg  = s0 + nt * 16 + (lane & 15);
    const int tg0 = t0 + wid * 16 + ((lane >> 4) << 3);
    float colp = 0.f;
#pragma unroll
    for (int v = 0; v < 8; ++v)
      colp += (sg <= tg0 + v) ? __expf(SCALE * d[v]) : 0.f;
    colp += __shfl_xor(colp, 16);            // fold wave32 halves (rows 0-7/8-15)
    if (lane < 16) atomicAdd(&dsum[nt * 16 + lane], colp);
  }
  __syncthreads();
  if (tid < 128) atomicAdd(&den[(size_t)b * SEQ + s0 + tid], dsum[tid]);
}

// ---------------------------------------------------------------------------
// Kernel 3: O[t][h] = sum_{s<=t} exp(scale*q.k) * (v[s][h]/den[s])
// One block per (batch, 128-row query tile); loop s in chunks of 32.
// ---------------------------------------------------------------------------
__global__ void __launch_bounds__(256) attn_out_kernel(
    const __bf16* __restrict__ qb, const __bf16* __restrict__ kb,
    const __bf16* __restrict__ vT, const float* __restrict__ den,
    float* __restrict__ out)
{
  constexpr int LDQ = 136, LDP = 40;
  __shared__ __align__(16) __bf16 Qs[128 * LDQ];   // q tile [t][h]
  __shared__ __align__(16) __bf16 Ks[32 * LDQ];    // k rows [s][h]
  __shared__ __align__(16) __bf16 Ps[128 * LDP];   // exp-scores [t][s]
  __shared__ __align__(16) __bf16 Vs[128 * LDP];   // v/den     [h][s]
  __shared__ float rden[32];
  const int tid  = threadIdx.x;
  const int lane = tid & 31;
  const int wid  = tid >> 5;
  const int b    = blockIdx.y;
  const int t0   = blockIdx.x * 128;

  const __bf16* qsrc = qb + (size_t)(b * SEQ + t0) * HEAD_D;
  const __bf16* ksrc = kb + (size_t)b * SEQ * HEAD_D;
  const __bf16* vsrc = vT + (size_t)b * HEAD_D * SEQ;
  const float*  denb = den + (size_t)b * SEQ;

  for (int i = tid; i < 128 * 16; i += 256) {        // stage Q once (async)
    const int r = i >> 4, c = (i & 15) << 3;
    g2s_b128(&Qs[r * LDQ + c], &qsrc[r * HEAD_D + c]);
  }
  async_wait0();
  __syncthreads();

  v16bf aq[4];                                       // Q frags live whole loop
#pragma unroll
  for (int h4 = 0; h4 < 4; ++h4)
    aq[h4] = load_a_frag(Qs + h4 * 32, LDQ, wid * 16, lane);

  v8f acc[8] = {};
  const int send = t0 + 128;
  for (int s0 = 0; s0 < send; s0 += 32) {
    __syncthreads();                                  // prev readers done
    for (int i = tid; i < 32 * 16; i += 256) {        // stage 32 k rows (async)
      const int r = i >> 4, c = (i & 15) << 3;
      g2s_b128(&Ks[r * LDQ + c], &ksrc[(size_t)(s0 + r) * HEAD_D + c]);
    }
    if (tid < 32) rden[tid] = 1.0f / denb[s0 + tid];
    async_wait0();
    __syncthreads();

    // stage V' = v/den into B-operand layout [h][s], 16B vector ops
    for (int i = tid; i < (128 * 32) / 8; i += 256) {
      const int h = i >> 2, c = (i & 3) << 3;
      const v8bf vv = *(const v8bf*)&vsrc[(size_t)h * SEQ + s0 + c];
      v8bf o;
#pragma unroll
      for (int j = 0; j < 8; ++j) o[j] = (__bf16)((float)vv[j] * rden[c + j]);
      *(v8bf*)&Vs[h * LDP + c] = o;
    }
    // exp-score tile P (16 t-rows per wave x 32 s-cols)
#pragma unroll
    for (int nt = 0; nt < 2; ++nt) {
      v8f d = {};
#pragma unroll
      for (int h4 = 0; h4 < 4; ++h4) {
        const v16bf bfr = load_b_frag(Ks + h4 * 32, LDQ, nt * 16, lane);
        d = wmma_bf16(aq[h4], bfr, d);
      }
      const int sl  = nt * 16 + (lane & 15);
      const int sg  = s0 + sl;
      const int tl0 = wid * 16 + ((lane >> 4) << 3);
#pragma unroll
      for (int v = 0; v < 8; ++v) {
        const float e = (sg <= t0 + tl0 + v) ? __expf(SCALE * d[v]) : 0.f;
        Ps[(tl0 + v) * LDP + sl] = (__bf16)e;
      }
    }
    __syncthreads();

    // O += P x V'
    const v16bf a = load_a_frag(Ps, LDP, wid * 16, lane);
#pragma unroll
    for (int ht = 0; ht < 8; ++ht) {
      const v16bf bfr = load_b_frag(Vs, LDP, ht * 16, lane);
      acc[ht] = wmma_bf16(a, bfr, acc[ht]);
    }
  }

#pragma unroll
  for (int ht = 0; ht < 8; ++ht) {
    const int col = ht * 16 + (lane & 15);
#pragma unroll
    for (int v = 0; v < 8; ++v) {
      const int row = wid * 16 + v + ((lane >> 4) << 3);
      out[(size_t)(b * SEQ + t0 + row) * HEAD_D + col] = acc[ht][v];
    }
  }
}

// ---------------------------------------------------------------------------
extern "C" void kernel_launch(void* const* d_in, const int* in_sizes, int n_in,
                              void* d_out, int out_size, void* d_ws, size_t ws_size,
                              hipStream_t stream) {
  const float* X  = (const float*)d_in[0];
  const float* Wk = (const float*)d_in[1];
  const float* bk = (const float*)d_in[2];
  const float* Wq = (const float*)d_in[3];
  const float* bq = (const float*)d_in[4];
  const float* Wv = (const float*)d_in[5];
  const float* bv = (const float*)d_in[6];
  float* out = (float*)d_out;

  const size_t qkvElems = (size_t)BATCH * SEQ * HEAD_D;
  __bf16* qb  = (__bf16*)d_ws;
  __bf16* kb  = qb + qkvElems;
  __bf16* vT  = kb + qkvElems;
  float*  den = (float*)(vT + qkvElems);

  (void)hipMemsetAsync(den, 0, (size_t)BATCH * SEQ * sizeof(float), stream);

  dim3 g1((BATCH * SEQ) / 128, 3);
  qkv_proj_kernel<<<g1, 256, 0, stream>>>(X, Wq, bq, Wk, bk, Wv, bv, qb, kb, vT);

  dim3 g2(136, BATCH);                 // 16*17/2 lower-triangular tile pairs
  colsum_kernel<<<g2, 256, 0, stream>>>(qb, kb, den);

  dim3 g3(SEQ / 128, BATCH);
  attn_out_kernel<<<g3, 256, 0, stream>>>(qb, kb, vT, den, out);
}
